// QuantizedLinear_41154376630555
// MI455X (gfx1250) — compile-verified, matrix operand order checked by reading, not run on hardware
//
#include <hip/hip_runtime.h>

typedef _Float16 v16h __attribute__((ext_vector_type(16)));
typedef _Float16 v8h  __attribute__((ext_vector_type(8)));
typedef _Float16 v4h  __attribute__((ext_vector_type(4)));
typedef float    v8f  __attribute__((ext_vector_type(8)));
typedef int      gv4i __attribute__((vector_size(16)));   // matches async builtin param type

#define M_TOT 8192
#define N_TOT 4096
#define K_TOT 4096
#define BM 128
#define BN 128
#define BK 64
#define LDT 72   // f16 elems per LDS row: 64 data + 8 pad (144B stride, bank-friendly)

// A tile: stride between a thread's consecutive loads (16 rows of K_TOT f32)
#define A_LD_STRIDE (16 * K_TOT * 4)   // 262144 B, fits signed 24-bit ioffset
// B tile: stride between a thread's consecutive chunks (64 rows of K_TOT i8)
#define B_LD_STRIDE (64 * K_TOT)       // 262144 B

#if __has_builtin(__builtin_amdgcn_global_load_async_to_lds_b128)
#define USE_ASYNC 1
#else
#define USE_ASYNC 0
#endif

__device__ __forceinline__ void wait_async0() {
#if __has_builtin(__builtin_amdgcn_s_wait_asynccnt)
    __builtin_amdgcn_s_wait_asynccnt(0);
#else
    asm volatile("s_wait_asynccnt 0x0" ::: "memory");
#endif
}

// Async loads complete in order: <=2 outstanding guarantees the OLDER batch of
// 2 copies has landed while the newest batch stays in flight (full overlap).
__device__ __forceinline__ void wait_async_le2() {
#if __has_builtin(__builtin_amdgcn_s_wait_asynccnt)
    __builtin_amdgcn_s_wait_asynccnt(2);
#else
    asm volatile("s_wait_asynccnt 0x2" ::: "memory");
#endif
}

__global__ __launch_bounds__(256) void qlinear_wmma_f16(
    const float* __restrict__ x,
    const signed char* __restrict__ qw,
    const float* __restrict__ wscale,
    const float* __restrict__ bias,
    float* __restrict__ out)
{
    __shared__ __align__(16) _Float16 As[BM * LDT];
    __shared__ __align__(16) _Float16 Bs[BN * LDT];
#if USE_ASYNC
    __shared__ __align__(16) signed char Bs8[2][BN * BK];  // raw int8 staging, double-buffered
#endif

    const int tid  = threadIdx.x;
    const int lane = tid & 31;
    const int wv   = tid >> 5;   // wave id 0..7
    const int wm   = wv & 3;     // 4 waves along M
    const int wn   = wv >> 2;    // 2 waves along N
    const int lr   = lane & 15;  // row within 16x16 tile
    const int hi   = lane >> 4;  // half-wave selector (K interleave)

    const int m0 = blockIdx.y * BM;
    const int n0 = blockIdx.x * BN;

    // Per-thread base pointers; advance by a constant each K-chunk so loads use
    // immediate instruction offsets instead of recomputed 64-bit adds.
    const char* pa = (const char*)(x  + (size_t)(m0 + (tid >> 4)) * K_TOT) + (tid & 15) * 16;
    const char* pb = (const char*)(qw + (size_t)(n0 + (tid >> 2)) * K_TOT) + (tid & 3) * 16;

    float4 ra[8];   // staged fp32 x tile: 128x64 f32 / 256 thr = 8 x float4
#if !USE_ASYNC
    int4   rb[2];   // staged int8 W tile: 128x64 i8 / 256 thr = 2 x 16B
#endif

    auto load_a = [&]() {
        #pragma unroll
        for (int i = 0; i < 8; ++i)
            ra[i] = *(const float4*)(pa + i * A_LD_STRIDE);
        pa += BK * 4;
    };

    auto load_b = [&](int buf) {
#if USE_ASYNC
        // Raw async byte-copy into LDS: LDS dest for chunk f = i*256+tid is
        // simply Bs8[buf] + f*16 (rows of 64B laid out contiguously).
        __builtin_amdgcn_global_load_async_to_lds_b128(
            (__attribute__((address_space(1))) gv4i*)(void*)pb,
            (__attribute__((address_space(3))) gv4i*)(void*)(&Bs8[buf][tid * 16]),
            0, 0);
        __builtin_amdgcn_global_load_async_to_lds_b128(
            (__attribute__((address_space(1))) gv4i*)(void*)(pb + B_LD_STRIDE),
            (__attribute__((address_space(3))) gv4i*)(void*)(&Bs8[buf][tid * 16 + 4096]),
            0, 0);
#else
        (void)buf;
        #pragma unroll
        for (int i = 0; i < 2; ++i)
            rb[i] = *(const int4*)(pb + i * B_LD_STRIDE);
#endif
        pb += BK;
    };

    auto stage = [&](int buf) {
        // A: f32 regs -> f16 LDS
        #pragma unroll
        for (int i = 0; i < 8; ++i) {
            int f = i * 256 + tid;
            int r = f >> 4, c = f & 15;
            float4 v = ra[i];
            v4h h = { (_Float16)v.x, (_Float16)v.y, (_Float16)v.z, (_Float16)v.w };
            *(v4h*)(&As[r * LDT + c * 4]) = h;          // ds_store_b64
        }
        // B: raw int8 -> f16 LDS (int8 is exact in f16)
        #pragma unroll
        for (int i = 0; i < 2; ++i) {
            int f = i * 256 + tid;
            int r = f >> 2, q = f & 3;
#if USE_ASYNC
            int4 d = *(const int4*)(&Bs8[buf][f * 16]); // ds_load_b128
#else
            (void)buf;
            int4 d = rb[i];
#endif
            const int dw[4] = { d.x, d.y, d.z, d.w };
            v8h h0, h1;
            #pragma unroll
            for (int j = 0; j < 8; ++j) {
                int b0 = (dw[j >> 2] << (24 - 8 * (j & 3))) >> 24;
                h0[j] = (_Float16)b0;
            }
            #pragma unroll
            for (int j = 8; j < 16; ++j) {
                int b0 = (dw[j >> 2] << (24 - 8 * (j & 3))) >> 24;
                h1[j - 8] = (_Float16)b0;
            }
            *(v8h*)(&Bs[r * LDT + q * 16])     = h0;    // ds_store_b128
            *(v8h*)(&Bs[r * LDT + q * 16 + 8]) = h1;
        }
    };

    v8f acc[2][4] = {};

    constexpr int T = K_TOT / BK;   // 64 K-chunks

    // Prologue: copies for tiles 0 and 1 in flight, then stage tile 0.
    load_b(0);
#if USE_ASYNC
    load_b(1);            // prefetch distance 2
    load_a();
    wait_async_le2();     // oldest batch (tile 0) complete; tile 1 still in flight
    __syncthreads();
#else
    load_a();
#endif
    stage(0);
    __syncthreads();

    for (int t = 0; t < T; ++t) {
#if USE_ASYNC
        if (t + 2 < T) load_b((t + 2) & 1);   // overlaps a FULL iteration of WMMA
#else
        if (t + 1 < T) load_b(0);
#endif
        if (t + 1 < T) load_a();

        #pragma unroll
        for (int kk = 0; kk < BK; kk += 32) {
            v16h aF[2], bF[4];
            // 16-bit A layout: lane<16 -> K0..7 & K16..23, lane>=16 -> +8
            #pragma unroll
            for (int im = 0; im < 2; ++im) {
                const _Float16* p = &As[(wm * 32 + im * 16 + lr) * LDT + kk + hi * 8];
                v8h lo = *(const v8h*)p;          // ds_load_b128
                v8h hh = *(const v8h*)(p + 16);   // ds_load_b128
                aF[im] = __builtin_shufflevector(lo, hh,
                        0,1,2,3,4,5,6,7,8,9,10,11,12,13,14,15);
            }
            // W stored [N,K] row-major -> B fragments use identical pattern
            #pragma unroll
            for (int in4 = 0; in4 < 4; ++in4) {
                const _Float16* p = &Bs[(wn * 64 + in4 * 16 + lr) * LDT + kk + hi * 8];
                v8h lo = *(const v8h*)p;
                v8h hh = *(const v8h*)(p + 16);
                bF[in4] = __builtin_shufflevector(lo, hh,
                        0,1,2,3,4,5,6,7,8,9,10,11,12,13,14,15);
            }
            #pragma unroll
            for (int im = 0; im < 2; ++im)
                #pragma unroll
                for (int in4 = 0; in4 < 4; ++in4)
                    acc[im][in4] = __builtin_amdgcn_wmma_f32_16x16x32_f16(
                        false, aF[im], false, bF[in4],
                        (short)0, acc[im][in4], false, false);
        }

#if USE_ASYNC
        if (t + 1 < T) {
            if (t + 2 < T) wait_async_le2();  // tile t+1 landed; t+2 in flight
            else           wait_async0();     // no newer batch: drain fully
        }
#endif
        __syncthreads();                 // As/Bs reads done + all async data visible
        if (t + 1 < T) { stage((t + 1) & 1); __syncthreads(); }
    }

    // Epilogue: D layout — VGPR r: lanes 0-15 -> M=r, lanes 16-31 -> M=8+r; N = lane&15
    const float s = wscale[0];
    #pragma unroll
    for (int in4 = 0; in4 < 4; ++in4) {
        const int ncol = n0 + wn * 64 + in4 * 16 + lr;
        const float bv = bias[ncol];
        #pragma unroll
        for (int im = 0; im < 2; ++im) {
            const int mrow = m0 + wm * 32 + im * 16 + hi * 8;
            float* op = out + (size_t)mrow * N_TOT + ncol;
            #pragma unroll
            for (int r = 0; r < 8; ++r)
                op[(size_t)r * N_TOT] = acc[im][in4][r] * s + bv;
        }
    }
}

extern "C" void kernel_launch(void* const* d_in, const int* in_sizes, int n_in,
                              void* d_out, int out_size, void* d_ws, size_t ws_size,
                              hipStream_t stream) {
    const float*       xp = (const float*)d_in[0];
    const signed char* wp = (const signed char*)d_in[1];
    const float*       sp = (const float*)d_in[2];
    const float*       bp = (const float*)d_in[3];
    float*             op = (float*)d_out;

    dim3 grid(N_TOT / BN, M_TOT / BM, 1);   // 32 x 64 workgroups
    qlinear_wmma_f16<<<grid, dim3(256, 1, 1), 0, stream>>>(xp, wp, sp, bp, op);

    (void)in_sizes; (void)n_in; (void)out_size; (void)d_ws; (void)ws_size;
}
